// PartSelfAttentionLayer_27315992002773
// MI455X (gfx1250) — matrix operand order repaired and checked by reading, not compiled, not run on hardware
//
#include <hip/hip_runtime.h>
#include <hip/hip_bf16.h>
#include <math.h>

// ---------------------------------------------------------------------------
// Problem constants (B=32, S=512, D=1024, H=16, HD=64)
// ---------------------------------------------------------------------------
#define BB   32
#define SS   512
#define DD   1024
#define HH   16
#define HDIM 64
#define BSTOK (BB * SS)      // 16384 tokens
#define DFF  2048

typedef _Float16 v16h __attribute__((ext_vector_type(16)));
typedef _Float16 v8h  __attribute__((ext_vector_type(8)));
typedef float    v8f  __attribute__((ext_vector_type(8)));

// GEMM epilogue modes
enum { M_QKV = 0, M_SCORES = 1, M_CTX = 2, M_BIAS = 3, M_GELU = 4 };

// ---------------------------------------------------------------------------
// WMMA wrapper: D = A(16x32 f16) * B(32x16 f16) + C(16x16 f32)
// ---------------------------------------------------------------------------
__device__ __forceinline__ v8f wmma_f16(v16h a, v16h b, v8f c) {
  return __builtin_amdgcn_wmma_f32_16x16x32_f16(
      /*neg_a=*/false, a, /*neg_b=*/false, b,
      /*c_mod=*/(short)0, c, /*reuse_a=*/false, /*reuse_b=*/false);
}

// ---------------------------------------------------------------------------
// A-fragment loader: 16x32 tile of row-major [rows, ld] matrix.
// ISA layout: lanes 0-15 (row r) hold K = k0..k0+7 (elems 0-7) and
// k0+16..k0+23 (elems 8-15); lanes 16-31 are shifted by +8 in K.
// ---------------------------------------------------------------------------
template <bool SRC_F32>
__device__ __forceinline__ v16h load_frag_a(const void* base, int row, int k0,
                                            int ld, int lane) {
  const int r  = row + (lane & 15);
  const int kc = k0 + ((lane >> 4) << 3);
  if constexpr (SRC_F32) {
    const float* p = (const float*)base + (size_t)r * ld + kc;
    v8f lo = *(const v8f*)(p);
    v8f hi = *(const v8f*)(p + 16);
    v16h f;
#pragma unroll
    for (int i = 0; i < 8; ++i) {
      f[i]     = (_Float16)lo[i];
      f[i + 8] = (_Float16)hi[i];
    }
    return f;
  } else {
    const _Float16* p = (const _Float16*)base + (size_t)r * ld + kc;
    v8h lo = *(const v8h*)(p);
    v8h hi = *(const v8h*)(p + 16);
    return __builtin_shufflevector(lo, hi, 0, 1, 2, 3, 4, 5, 6, 7, 8, 9, 10,
                                   11, 12, 13, 14, 15);
  }
}

// ---------------------------------------------------------------------------
// B-fragment loader: B[k][n] = W[n][k], W row-major [N, ld].
// ISA layout: lane L holds column n = L&15; lanes 0-15 hold K = k0..k0+7
// and k0+16..k0+23; lanes 16-31 are shifted by +8 in K (matches A).
// ---------------------------------------------------------------------------
__device__ __forceinline__ v16h load_frag_b(const void* base, int nrow, int k0,
                                            int ld, int lane) {
  const int r  = nrow + (lane & 15);
  const int kc = k0 + ((lane >> 4) << 3);
  const _Float16* p = (const _Float16*)base + (size_t)r * ld + kc;
  v8h lo = *(const v8h*)(p);
  v8h hi = *(const v8h*)(p + 16);
  return __builtin_shufflevector(lo, hi, 0, 1, 2, 3, 4, 5, 6, 7, 8, 9, 10, 11,
                                 12, 13, 14, 15);
}

// ---------------------------------------------------------------------------
// Generic WMMA GEMM:  C[M,N] = A[M,K] * W[N,K]^T  (+ mode-specific epilogue)
// Block = 128 threads = 4 waves in 2x2.
// Wave tile: 32 x (16*NT); block tile: 64 x (32*NT).  NT = 4 (or 2 for ctx).
// blockIdx.z selects a (b,h) slice via byte strides. W operand is always f16.
// ---------------------------------------------------------------------------
template <int MODE, bool AF32, int NT>
__global__ __launch_bounds__(128) void gemm_wmma(
    const void* __restrict__ A, const void* __restrict__ Wm,
    const float* __restrict__ bias, void* __restrict__ out,
    _Float16* __restrict__ p0, _Float16* __restrict__ p1,
    _Float16* __restrict__ p2, int M, int N, int K, int lda, int ldw,
    size_t strideAbytes, size_t strideWbytes) {
  const int lane  = threadIdx.x & 31;
  const int w     = threadIdx.x >> 5;
  const int mBase = blockIdx.x * 64 + (w >> 1) * 32;
  const int nBase = blockIdx.y * (32 * NT) + (w & 1) * (16 * NT);
  const int z     = blockIdx.z;

  const char* Ab = (const char*)A + (size_t)z * strideAbytes;
  const char* Wb = (const char*)Wm + (size_t)z * strideWbytes;

  v8f acc[2][NT] = {};

  for (int k0 = 0; k0 < K; k0 += 32) {
    v16h a0 = load_frag_a<AF32>(Ab, mBase, k0, lda, lane);
    v16h a1 = load_frag_a<AF32>(Ab, mBase + 16, k0, lda, lane);
    v16h bfr[NT];
#pragma unroll
    for (int ni = 0; ni < NT; ++ni)
      bfr[ni] = load_frag_b(Wb, nBase + ni * 16, k0, ldw, lane);
#pragma unroll
    for (int ni = 0; ni < NT; ++ni) {
      acc[0][ni] = wmma_f16(a0, bfr[ni], acc[0][ni]);
      acc[1][ni] = wmma_f16(a1, bfr[ni], acc[1][ni]);
    }
  }

  // C layout: element v of lane L -> row (v + 8*(L>=16)), col (L%16)
  const int rsel = (lane >> 4) << 3;
  const int csel = lane & 15;

#pragma unroll
  for (int mi = 0; mi < 2; ++mi) {
#pragma unroll
    for (int ni = 0; ni < NT; ++ni) {
#pragma unroll
      for (int v = 0; v < 8; ++v) {
        const int m = mBase + mi * 16 + v + rsel;
        const int n = nBase + ni * 16 + csel;
        float val   = acc[mi][ni][v];

        if constexpr (MODE == M_QKV) {
          val += bias[n];
          const int b = m >> 9;      // token / S
          const int s = m & (SS - 1);
          if (n < DD) {
            const int hh = n >> 6, d = n & (HDIM - 1);
            p0[((((size_t)b * HH) + hh) * SS + s) * HDIM + d] =
                (_Float16)(val * 0.125f);  // q * 1/sqrt(64)
          } else if (n < 2 * DD) {
            const int n2 = n - DD;
            const int hh = n2 >> 6, d = n2 & (HDIM - 1);
            p1[((((size_t)b * HH) + hh) * SS + s) * HDIM + d] = (_Float16)val;
          } else {
            const int n2 = n - 2 * DD;
            const int hh = n2 >> 6, d = n2 & (HDIM - 1);
            // v stored transposed: vT[b][h][d][s]
            p2[((((size_t)b * HH) + hh) * HDIM + d) * SS + s] = (_Float16)val;
          }
        } else if constexpr (MODE == M_SCORES) {
          ((float*)out)[(size_t)z * M * N + (size_t)m * N + n] = val;
        } else if constexpr (MODE == M_CTX) {
          const int b = z >> 4, hh = z & (HH - 1);
          ((_Float16*)out)[((size_t)((b << 9) + m)) * DD + hh * HDIM + n] =
              (_Float16)val;
        } else if constexpr (MODE == M_BIAS) {
          ((float*)out)[(size_t)m * N + n] = val + bias[n];
        } else {  // M_GELU (exact erf)
          const float t = val + bias[n];
          ((_Float16*)out)[(size_t)m * N + n] =
              (_Float16)(0.5f * t * (1.0f + erff(t * 0.70710678118654752f)));
        }
      }
    }
  }
}

// ---------------------------------------------------------------------------
// Elementwise f32 -> f16 cast, 8 elems/thread.
// ---------------------------------------------------------------------------
__global__ __launch_bounds__(256) void cast_f32_f16(const float* __restrict__ in,
                                                    _Float16* __restrict__ out,
                                                    int n8) {
  const int idx = blockIdx.x * 256 + threadIdx.x;
  if (idx >= n8) return;
  const size_t base = (size_t)idx * 8;
  v8f v = *(const v8f*)(in + base);
  v8h o;
#pragma unroll
  for (int i = 0; i < 8; ++i) o[i] = (_Float16)v[i];
  *(v8h*)(out + base) = o;
}

// ---------------------------------------------------------------------------
// In-place row softmax over the attn_weights region (rows of 512 f32).
// One block (128 threads, 4 elems each) per row.
// ---------------------------------------------------------------------------
__global__ __launch_bounds__(128) void softmax_rows(float* __restrict__ attn) {
  __shared__ float red[128];
  float* p     = attn + (size_t)blockIdx.x * SS;
  const int t  = threadIdx.x;
  float x0 = p[t], x1 = p[t + 128], x2 = p[t + 256], x3 = p[t + 384];

  float mx = fmaxf(fmaxf(x0, x1), fmaxf(x2, x3));
  red[t] = mx;
  __syncthreads();
  for (int s = 64; s > 0; s >>= 1) {
    if (t < s) red[t] = fmaxf(red[t], red[t + s]);
    __syncthreads();
  }
  mx = red[0];
  __syncthreads();

  const float e0 = expf(x0 - mx), e1 = expf(x1 - mx);
  const float e2 = expf(x2 - mx), e3 = expf(x3 - mx);
  red[t] = e0 + e1 + e2 + e3;
  __syncthreads();
  for (int s = 64; s > 0; s >>= 1) {
    if (t < s) red[t] += red[t + s];
    __syncthreads();
  }
  const float inv = 1.0f / red[0];

  p[t] = e0 * inv;
  p[t + 128] = e1 * inv;
  p[t + 256] = e2 * inv;
  p[t + 384] = e3 * inv;
}

// ---------------------------------------------------------------------------
// h = LayerNorm(x + r) * w + b  over rows of 1024 f32.
// One block (256 threads, 4 elems each) per row. Optional f16 mirror output.
// ---------------------------------------------------------------------------
__device__ __forceinline__ float block_sum(float v, float* red) {
  const int t = threadIdx.x;
  red[t] = v;
  __syncthreads();
  for (int s = 128; s > 0; s >>= 1) {
    if (t < s) red[t] += red[t + s];
    __syncthreads();
  }
  const float r = red[0];
  __syncthreads();
  return r;
}

__global__ __launch_bounds__(256) void ln_residual(
    const float* __restrict__ x, const float* __restrict__ r,
    const float* __restrict__ w, const float* __restrict__ b,
    float* __restrict__ out, _Float16* __restrict__ out16) {
  __shared__ float red[256];
  const size_t row = blockIdx.x;
  const float* px  = x + row * DD;
  const float* pr  = r + row * DD;
  float* po        = out + row * DD;
  const int t      = threadIdx.x;

  float v[4];
  float s = 0.f;
#pragma unroll
  for (int j = 0; j < 4; ++j) {
    v[j] = px[t + j * 256] + pr[t + j * 256];
    s += v[j];
  }
  const float mu = block_sum(s, red) * (1.0f / DD);

  float s2 = 0.f;
#pragma unroll
  for (int j = 0; j < 4; ++j) {
    const float d = v[j] - mu;
    s2 += d * d;
  }
  const float var = block_sum(s2, red) * (1.0f / DD);
  const float rs  = rsqrtf(var + 1e-5f);

#pragma unroll
  for (int j = 0; j < 4; ++j) {
    const int i   = t + j * 256;
    const float o = (v[j] - mu) * rs * w[i] + b[i];
    po[i] = o;
    if (out16) out16[row * DD + i] = (_Float16)o;
  }
}

// ---------------------------------------------------------------------------
// Launch
// ---------------------------------------------------------------------------
extern "C" void kernel_launch(void* const* d_in, const int* in_sizes, int n_in,
                              void* d_out, int out_size, void* d_ws,
                              size_t ws_size, hipStream_t stream) {
  (void)in_sizes; (void)n_in; (void)out_size; (void)ws_size;

  const float* x         = (const float*)d_in[0];   // [B,S,D]
  const float* in_proj_w = (const float*)d_in[1];   // [3D,D]
  const float* in_proj_b = (const float*)d_in[2];   // [3D]
  const float* out_w     = (const float*)d_in[3];   // [D,D]
  const float* out_b     = (const float*)d_in[4];   // [D]
  const float* ln1_w     = (const float*)d_in[5];
  const float* ln1_b     = (const float*)d_in[6];
  const float* w1        = (const float*)d_in[7];   // [2D,D]
  const float* b1        = (const float*)d_in[8];
  const float* w2        = (const float*)d_in[9];   // [D,2D]
  const float* b2        = (const float*)d_in[10];
  const float* ln2_w     = (const float*)d_in[11];
  const float* ln2_b     = (const float*)d_in[12];

  // d_out = [h2 (B*S*D f32)][attn_weights (B*H*S*S f32)]
  float* h2_out   = (float*)d_out;
  float* attn_out = (float*)d_out + (size_t)BSTOK * DD;

  // Workspace layout (~386 MB):
  char* ws = (char*)d_ws;
  _Float16* q16    = (_Float16*)(ws);                    // 33.5 MB
  _Float16* k16    = (_Float16*)(ws + 33554432ull);      // 33.5 MB
  _Float16* vT16   = (_Float16*)(ws + 67108864ull);      // 33.5 MB
  float*    h32    = (float*)(ws + 100663296ull);        // 67 MB
  float*    ao32   = (float*)(ws + 167772160ull);        // 67 MB (attn_out/ff2)
  _Float16* ff16   = (_Float16*)(ws + 234881024ull);     // 67 MB
  _Float16* x16    = (_Float16*)(ws + 301989888ull);     // 33.5 MB
  _Float16* h16    = (_Float16*)(ws + 335544320ull);     // 33.5 MB
  _Float16* wqkv16 = (_Float16*)(ws + 369098752ull);     // 6 MB
  _Float16* wout16 = (_Float16*)(ws + 375390208ull);     // 2 MB
  _Float16* w116   = (_Float16*)(ws + 377487360ull);     // 4 MB
  _Float16* w216   = (_Float16*)(ws + 381681664ull);     // 4 MB
  _Float16* ctx16  = q16;  // q consumed before ctx is written

  const dim3 blk(128);

  // 0) one-time (per launch) f32->f16 casts of reused operands
  cast_f32_f16<<<(BSTOK * DD / 8) / 256, 256, 0, stream>>>(x, x16,
                                                           BSTOK * DD / 8);
  cast_f32_f16<<<(3 * DD * DD / 8) / 256, 256, 0, stream>>>(
      in_proj_w, wqkv16, 3 * DD * DD / 8);
  cast_f32_f16<<<(DD * DD / 8) / 256, 256, 0, stream>>>(out_w, wout16,
                                                        DD * DD / 8);
  cast_f32_f16<<<(DFF * DD / 8) / 256, 256, 0, stream>>>(w1, w116,
                                                         DFF * DD / 8);
  cast_f32_f16<<<(DD * DFF / 8) / 256, 256, 0, stream>>>(w2, w216,
                                                         DD * DFF / 8);

  // 1) QKV projection: [16384,3072] = x16 @ wqkv16^T  (split q/k/vT epilogue)
  gemm_wmma<M_QKV, false, 4><<<dim3(BSTOK / 64, (3 * DD) / 128, 1), blk, 0,
                               stream>>>(
      x16, wqkv16, in_proj_b, nullptr, q16, k16, vT16, BSTOK, 3 * DD, DD, DD,
      DD, 0, 0);

  // 2) scores per (b,h): [512,512] = q[512,64] @ k[512,64]^T -> attn region
  gemm_wmma<M_SCORES, false, 4><<<dim3(SS / 64, SS / 128, BB * HH), blk, 0,
                                  stream>>>(
      q16, k16, nullptr, attn_out, nullptr, nullptr, nullptr, SS, SS, HDIM,
      HDIM, HDIM, (size_t)SS * HDIM * 2, (size_t)SS * HDIM * 2);

  // 3) softmax in place over attn region (this IS output #2)
  softmax_rows<<<BB * HH * SS, 128, 0, stream>>>(attn_out);

  // 4) ctx per (b,h): [512,64] = P[512,512](f32) @ vT[64,512]^T -> ctx16
  gemm_wmma<M_CTX, true, 2><<<dim3(SS / 64, HDIM / 64, BB * HH), blk, 0,
                              stream>>>(
      attn_out, vT16, nullptr, ctx16, nullptr, nullptr, nullptr, SS, HDIM, SS,
      SS, SS, (size_t)SS * SS * 4, (size_t)HDIM * SS * 2);

  // 5) out projection: ao32 = ctx16 @ wout16^T + out_b
  gemm_wmma<M_BIAS, false, 4><<<dim3(BSTOK / 64, DD / 128, 1), blk, 0,
                                stream>>>(
      ctx16, wout16, out_b, ao32, nullptr, nullptr, nullptr, BSTOK, DD, DD, DD,
      DD, 0, 0);

  // 6) h = LN1(x + attn_out) -> h32 (f32) and h16 (f16 for FFN1)
  ln_residual<<<BSTOK, 256, 0, stream>>>(x, ao32, ln1_w, ln1_b, h32, h16);

  // 7) ff = gelu(h16 @ w116^T + b1) -> f16
  gemm_wmma<M_GELU, false, 4><<<dim3(BSTOK / 64, DFF / 128, 1), blk, 0,
                                stream>>>(
      h16, w116, b1, ff16, nullptr, nullptr, nullptr, BSTOK, DFF, DD, DD, DD,
      0, 0);

  // 8) ff2 = ff16 @ w216^T + b2 -> ao32 (reused)
  gemm_wmma<M_BIAS, false, 4><<<dim3(BSTOK / 64, DD / 128, 1), blk, 0,
                                stream>>>(
      ff16, w216, b2, ao32, nullptr, nullptr, nullptr, BSTOK, DD, DFF, DFF,
      DFF, 0, 0);

  // 9) h2 = LN2(h + ff2) -> output #1
  ln_residual<<<BSTOK, 256, 0, stream>>>(h32, ao32, ln2_w, ln2_b, h2_out,
                                         nullptr);
}
